// Block_20486994002331
// MI455X (gfx1250) — matrix operand order repaired and checked by reading, not compiled, not run on hardware
//
#include <hip/hip_runtime.h>
#include <hip/hip_bf16.h>
#include <math.h>
#include <stdint.h>

typedef __attribute__((ext_vector_type(16))) _Float16 v16h;
typedef __attribute__((ext_vector_type(8)))  float    v8f;

#define DIM   1024
#define NHEAD 16
#define NKV   4
#define HD    64
#define INTER 2688
#define BB    2
#define NN    2048
#define ROWS  (BB*NN)   // 4096

__device__ __forceinline__ v16h pack16(uint4 lo, uint4 hi) {
  union { uint4 u[2]; v16h v; } t;
  t.u[0] = lo; t.u[1] = hi;
  return t.v;
}

// ---------------------------------------------------------------- cast
__global__ void cast_f32_f16(const float* __restrict__ in, _Float16* __restrict__ out, int n) {
  int i = blockIdx.x * blockDim.x + threadIdx.x;
  if (i < n) out[i] = (_Float16)in[i];
}

// ---------------------------------------------------------------- rmsnorm -> f16
__global__ __launch_bounds__(256) void rmsnorm_f16(const float* __restrict__ x,
                                                   const float* __restrict__ w,
                                                   _Float16* __restrict__ out) {
  int row = blockIdx.x;
  const float* xr = x + (size_t)row * DIM;
  __shared__ float red[8];
  float s = 0.f;
  for (int i = threadIdx.x; i < DIM; i += 256) { float v = xr[i]; s += v * v; }
  for (int off = 16; off > 0; off >>= 1) s += __shfl_xor(s, off, 32);
  if ((threadIdx.x & 31) == 0) red[threadIdx.x >> 5] = s;
  __syncthreads();
  if (threadIdx.x == 0) {
    float tot = 0.f;
    for (int i = 0; i < 8; ++i) tot += red[i];
    red[0] = rsqrtf(tot / (float)DIM + 1e-6f);
  }
  __syncthreads();
  float r = red[0];
  for (int i = threadIdx.x; i < DIM; i += 256)
    out[(size_t)row * DIM + i] = (_Float16)(xr[i] * r * w[i]);
}

// ---------------------------------------------------------------- WMMA GEMM
// C[M,N] = A[M,K]*B[K,N] (f16 in, f32 acc). Workgroup tile 64x128, 8 waves
// (2M x 4N), wave tile 32x32, K step 32, double-buffered LDS (1 barrier/step).
// A tile is staged with the CDNA5 async global->LDS DMA (ASYNCcnt), waited
// only at the publishing barrier so the copy overlaps the WMMA stream.
// MODE: 0 = f32 store, 1 = f32 store + addend, 2 = f16 store.
template <int MODE>
__global__ __launch_bounds__(256) void gemm_wmma(
    const _Float16* __restrict__ A, const _Float16* __restrict__ Bm,
    float* __restrict__ Cf, _Float16* __restrict__ Ch,
    const float* __restrict__ addend, int M, int N, int K)
{
  __shared__ __align__(16) _Float16 sA[2][64][32];    // row major [m][k]
  __shared__ __align__(16) _Float16 sBT[2][128][32];  // transposed [n][k]
  int tid  = threadIdx.x;
  int wave = tid >> 5, lane = tid & 31;
  int hs   = lane >> 4, l16 = lane & 15;
  int m_blk = blockIdx.y * 64;
  int n_blk = blockIdx.x * 128;
  int wm = (wave >> 2) * 32;
  int wn = (wave & 3) * 32;

  v8f z8 = {0.f,0.f,0.f,0.f,0.f,0.f,0.f,0.f};
  v8f acc[2][2];
  acc[0][0] = z8; acc[0][1] = z8; acc[1][0] = z8; acc[1][1] = z8;

  int arow = tid >> 2;
  int acol = (tid & 3) * 8;

  auto stage = [&](int k0, int buf) {
    // A tile: one async 16B/lane global->LDS DMA (no VGPR round trip)
    {
      unsigned ldsa = (unsigned)(size_t)&sA[buf][arow][acol];
      const _Float16* gpa = A + (size_t)(m_blk + arow) * K + k0 + acol;
      asm volatile("global_load_async_to_lds_b128 %0, %1, off"
                   :: "v"(ldsa), "v"(gpa) : "memory");
    }
    // B tile: read row-major coalesced, scatter transposed into LDS
    #pragma unroll
    for (int c = 0; c < 2; ++c) {
      int idx = tid * 2 + c;           // 0..511 chunks of 8 halfs
      int kr  = idx >> 4;              // 0..31
      int nc  = (idx & 15) * 8;        // 0..120
      union { uint4 u; _Float16 h[8]; } tmp;
      tmp.u = *(const uint4*)(Bm + (size_t)(k0 + kr) * N + n_blk + nc);
      #pragma unroll
      for (int j = 0; j < 8; ++j) sBT[buf][nc + j][kr] = tmp.h[j];
    }
  };

  int nk = K >> 5;
  stage(0, 0);
  asm volatile("s_wait_asynccnt 0" ::: "memory");
  __syncthreads();

  for (int kb = 0; kb < nk; ++kb) {
    int cur = kb & 1;
    if (kb + 1 < nk) stage((kb + 1) << 5, cur ^ 1);
    if (kb + 2 < nk) {
      __builtin_prefetch(A + (size_t)(m_blk + arow) * K + ((kb + 2) << 5) + acol, 0, 3);
      __builtin_prefetch(Bm + (size_t)(((kb + 2) << 5) + (tid >> 4)) * N + n_blk + (tid & 15) * 8, 0, 3);
    }

    v16h af[2], bf[2];
    #pragma unroll
    for (int mt = 0; mt < 2; ++mt) {
      // A 16x32 layout: lanes<16 k0..7 & k16..23 ; lanes>=16 k8..15 & k24..31
      const _Float16* ar = &sA[cur][wm + mt * 16 + l16][hs * 8];
      af[mt] = pack16(*(const uint4*)ar, *(const uint4*)(ar + 16));
    }
    #pragma unroll
    for (int nt = 0; nt < 2; ++nt) {
      // B 32x16 layout: lanes<16 k0..7 & k16..23 ; lanes>=16 k8..15 & k24..31 (col = l16)
      const _Float16* br = &sBT[cur][wn + nt * 16 + l16][hs * 8];
      bf[nt] = pack16(*(const uint4*)br, *(const uint4*)(br + 16));
    }
    #pragma unroll
    for (int mt = 0; mt < 2; ++mt)
      #pragma unroll
      for (int nt = 0; nt < 2; ++nt)
        acc[mt][nt] = __builtin_amdgcn_wmma_f32_16x16x32_f16(
            false, af[mt], false, bf[nt], (short)0, acc[mt][nt], false, false);
    asm volatile("s_wait_asynccnt 0" ::: "memory");
    __syncthreads();
  }

  #pragma unroll
  for (int mt = 0; mt < 2; ++mt)
    #pragma unroll
    for (int nt = 0; nt < 2; ++nt)
      #pragma unroll
      for (int g = 0; g < 8; ++g) {
        int row = m_blk + wm + mt * 16 + 8 * hs + g;
        int col = n_blk + wn + nt * 16 + l16;
        size_t idx = (size_t)row * N + col;
        float v = acc[mt][nt][g];
        if (MODE == 1) v += addend[idx];
        if (MODE == 2) Ch[idx] = (_Float16)v;
        else           Cf[idx] = v;
      }
}

// ---------------------------------------------------------------- RoPE Q: [B,N,H,64] f32 -> [B,H,N,64] f16
__global__ void rope_q(const float* __restrict__ q32, const float* __restrict__ cb,
                       const float* __restrict__ sb, _Float16* __restrict__ q16) {
  int i = blockIdx.x * blockDim.x + threadIdx.x;      // BB*NN*NHEAD*32
  if (i >= BB * NN * NHEAD * 32) return;
  int p = i & 31;
  int h = (i >> 5) & 15;
  int n = (i >> 9) & (NN - 1);
  int b = i >> 20;
  size_t src = (((size_t)(b * NN + n)) * NHEAD + h) * HD;
  float xr = q32[src + 2 * p];
  float xi = q32[src + 2 * p + 1];
  float c = cb[n * 32 + p], s = sb[n * 32 + p];
  size_t dst = (((size_t)(b * NHEAD + h)) * NN + n) * HD;
  q16[dst + 2 * p]     = (_Float16)(xr * c - xi * s);
  q16[dst + 2 * p + 1] = (_Float16)(xr * s + xi * c);
}

// ---------------------------------------------------------------- RoPE K + copy V: kv [B,N,512] f32
__global__ void rope_kv(const float* __restrict__ kv32, const float* __restrict__ cb,
                        const float* __restrict__ sb,
                        _Float16* __restrict__ k16, _Float16* __restrict__ v16o) {
  int i = blockIdx.x * blockDim.x + threadIdx.x;      // BB*NN*NKV*32
  if (i >= BB * NN * NKV * 32) return;
  int p  = i & 31;
  int kh = (i >> 5) & 3;
  int n  = (i >> 7) & (NN - 1);
  int b  = i >> 18;
  const float* row = kv32 + (size_t)(b * NN + n) * (2 * NKV * HD);
  float xr = row[kh * HD + 2 * p];
  float xi = row[kh * HD + 2 * p + 1];
  float vr = row[NKV * HD + kh * HD + 2 * p];
  float vi = row[NKV * HD + kh * HD + 2 * p + 1];
  float c = cb[n * 32 + p], s = sb[n * 32 + p];
  size_t dst = ((size_t)(b * NKV + kh) * NN + n) * HD;
  k16[dst + 2 * p]      = (_Float16)(xr * c - xi * s);
  k16[dst + 2 * p + 1]  = (_Float16)(xr * s + xi * c);
  v16o[dst + 2 * p]     = (_Float16)vr;
  v16o[dst + 2 * p + 1] = (_Float16)vi;
}

// ---------------------------------------------------------------- flash attention
// 1 wave per 16 query rows, 32 keys per iteration -> all WMMAs fully dense.
__global__ __launch_bounds__(32) void flash_attn(const _Float16* __restrict__ q16,
                                                 const _Float16* __restrict__ k16,
                                                 const _Float16* __restrict__ v16,
                                                 _Float16* __restrict__ o16) {
  __shared__ __align__(16) _Float16 sP[16][32];   // P tile (16 q rows x 32 keys)
  __shared__ __align__(16) _Float16 sVT[64][32];  // V tile transposed [d][k]
  int bh = blockIdx.x;                  // b*16 + h
  int qt = blockIdx.y;                  // query 16-row tile
  int b = bh >> 4, h = bh & 15, kvh = h & 3;   // jnp.tile => head h uses kv head h%4
  int lane = threadIdx.x;
  int hs = lane >> 4, l16 = lane & 15;

  const _Float16* Q = q16 + ((size_t)bh * NN + qt * 16) * HD;
  const _Float16* K = k16 + ((size_t)(b * NKV + kvh) * NN) * HD;
  const _Float16* V = v16 + ((size_t)(b * NKV + kvh) * NN) * HD;

  // Q A-fragments for the two 32-wide d chunks
  v16h qa[2];
  #pragma unroll
  for (int c = 0; c < 2; ++c) {
    const _Float16* base = Q + l16 * HD + c * 32 + hs * 8;
    qa[c] = pack16(*(const uint4*)base, *(const uint4*)(base + 16));
  }

  float mrow[8], lrow[8];
  #pragma unroll
  for (int g = 0; g < 8; ++g) { mrow[g] = -3.0e38f; lrow[g] = 0.f; }
  v8f z8 = {0.f,0.f,0.f,0.f,0.f,0.f,0.f,0.f};
  v8f oacc[4];
  #pragma unroll
  for (int t = 0; t < 4; ++t) oacc[t] = z8;

  int ntiles = (qt >> 1) + 1;   // 32-key tiles; overhang is causally masked (max key 2047)
  for (int kt2 = 0; kt2 < ntiles; ++kt2) {
    int k0 = kt2 * 32;
    // S = Q K^T for two 16-key subtiles
    v8f sacc[2];
    #pragma unroll
    for (int j = 0; j < 2; ++j) {
      const _Float16* Kt = K + (size_t)(k0 + j * 16) * HD;
      v16h kb0, kb1;
      {
        const _Float16* base = Kt + l16 * HD + 0 * 32 + hs * 8;
        kb0 = pack16(*(const uint4*)base, *(const uint4*)(base + 16));
      }
      {
        const _Float16* base = Kt + l16 * HD + 1 * 32 + hs * 8;
        kb1 = pack16(*(const uint4*)base, *(const uint4*)(base + 16));
      }
      v8f s = z8;
      s = __builtin_amdgcn_wmma_f32_16x16x32_f16(false, qa[0], false, kb0, (short)0, s, false, false);
      s = __builtin_amdgcn_wmma_f32_16x16x32_f16(false, qa[1], false, kb1, (short)0, s, false, false);
      sacc[j] = s;
    }

    // online softmax over the 32 keys
    float alpha[8], p0[8], p1[8];
    #pragma unroll
    for (int g = 0; g < 8; ++g) {
      int qr  = qt * 16 + g + 8 * hs;
      int kr0 = k0 + l16;
      int kr1 = k0 + 16 + l16;
      float v0 = sacc[0][g] * 0.125f;            // 1/sqrt(64)
      float v1 = sacc[1][g] * 0.125f;
      if (kr0 > qr) v0 = -3.0e38f;               // causal mask
      if (kr1 > qr) v1 = -3.0e38f;
      float m = fmaxf(v0, v1);
      m = fmaxf(m, __shfl_xor(m, 1, 32));
      m = fmaxf(m, __shfl_xor(m, 2, 32));
      m = fmaxf(m, __shfl_xor(m, 4, 32));
      m = fmaxf(m, __shfl_xor(m, 8, 32));
      float mnew = fmaxf(mrow[g], m);
      alpha[g] = __expf(mrow[g] - mnew);
      mrow[g] = mnew;
      float e0 = __expf(v0 - mnew);
      float e1 = __expf(v1 - mnew);
      p0[g] = e0; p1[g] = e1;
      float rs2 = e0 + e1;
      rs2 += __shfl_xor(rs2, 1, 32);
      rs2 += __shfl_xor(rs2, 2, 32);
      rs2 += __shfl_xor(rs2, 4, 32);
      rs2 += __shfl_xor(rs2, 8, 32);
      lrow[g] = lrow[g] * alpha[g] + rs2;
    }
    // P (C layout) -> LDS row major [16][32]
    #pragma unroll
    for (int g = 0; g < 8; ++g) {
      sP[g + 8 * hs][l16]      = (_Float16)p0[g];
      sP[g + 8 * hs][16 + l16] = (_Float16)p1[g];
    }
    // V tile (32 keys x 64 d) transposed into LDS
    const _Float16* Vt = V + (size_t)k0 * HD;
    for (int idx = lane; idx < 256; idx += 32) {
      int kr = idx >> 3, d0 = (idx & 7) * 8;
      union { uint4 u; _Float16 hh[8]; } tmp;
      tmp.u = *(const uint4*)(Vt + kr * HD + d0);
      #pragma unroll
      for (int j = 0; j < 8; ++j) sVT[d0 + j][kr] = tmp.hh[j];
    }
    __syncthreads();

    // P as full 16x32 A-fragment
    const _Float16* pr = &sP[l16][hs * 8];
    v16h pa = pack16(*(const uint4*)pr, *(const uint4*)(pr + 16));
    #pragma unroll
    for (int t = 0; t < 4; ++t) {
      const _Float16* vr = &sVT[16 * t + l16][hs * 8];
      v16h vb = pack16(*(const uint4*)vr, *(const uint4*)(vr + 16));
      #pragma unroll
      for (int g = 0; g < 8; ++g) oacc[t][g] = oacc[t][g] * alpha[g];
      oacc[t] = __builtin_amdgcn_wmma_f32_16x16x32_f16(false, pa, false, vb, (short)0, oacc[t], false, false);
    }
    __syncthreads();
  }

  #pragma unroll
  for (int t = 0; t < 4; ++t)
    #pragma unroll
    for (int g = 0; g < 8; ++g) {
      int row = qt * 16 + g + 8 * hs;
      int d = 16 * t + l16;
      float v = oacc[t][g] / lrow[g];
      o16[((size_t)b * NN + row) * DIM + h * HD + d] = (_Float16)v;
    }
}

// ---------------------------------------------------------------- SwiGLU gate (in place into g1)
__global__ void swiglu_gate(_Float16* __restrict__ g1, const _Float16* __restrict__ g3, int n) {
  int i = blockIdx.x * blockDim.x + threadIdx.x;
  if (i >= n) return;
  float a = (float)g1[i], bgt = (float)g3[i];
  float s = a / (1.f + __expf(-a));
  g1[i] = (_Float16)(s * bgt);
}

// ================================================================ launch
extern "C" void kernel_launch(void* const* d_in, const int* in_sizes, int n_in,
                              void* d_out, int out_size, void* d_ws, size_t ws_size,
                              hipStream_t stream) {
  (void)in_sizes; (void)n_in; (void)out_size; (void)ws_size;
  const float* x   = (const float*)d_in[0];
  const float* wan = (const float*)d_in[1];
  const float* wfn = (const float*)d_in[2];
  const float* Wq  = (const float*)d_in[3];
  const float* Wkv = (const float*)d_in[4];
  const float* Wo  = (const float*)d_in[5];
  const float* W1  = (const float*)d_in[6];
  const float* W2  = (const float*)d_in[7];
  const float* W3  = (const float*)d_in[8];
  const float* rc  = (const float*)d_in[9];
  const float* rsn = (const float*)d_in[10];
  float* out = (float*)d_out;

  char* ws = (char*)d_ws;
  size_t off = 0;
  auto alloc = [&](size_t bytes) -> void* {
    void* p = ws + off;
    off += bytes;
    off = (off + 255) & ~(size_t)255;
    return p;
  };

  _Float16* Wq16  = (_Float16*)alloc((size_t)DIM * DIM * 2);
  _Float16* Wkv16 = (_Float16*)alloc((size_t)DIM * 512 * 2);
  _Float16* Wo16  = (_Float16*)alloc((size_t)DIM * DIM * 2);
  _Float16* W1h   = (_Float16*)alloc((size_t)DIM * INTER * 2);
  _Float16* W3h   = (_Float16*)alloc((size_t)DIM * INTER * 2);
  _Float16* W2h   = (_Float16*)alloc((size_t)INTER * DIM * 2);
  _Float16* h16   = (_Float16*)alloc((size_t)ROWS * DIM * 2);
  float*    q32   = (float*)alloc((size_t)ROWS * DIM * 4);
  float*    kv32  = (float*)alloc((size_t)ROWS * 512 * 4);
  _Float16* q16   = (_Float16*)alloc((size_t)BB * NHEAD * NN * HD * 2);
  _Float16* k16   = (_Float16*)alloc((size_t)BB * NKV * NN * HD * 2);
  _Float16* v16   = (_Float16*)alloc((size_t)BB * NKV * NN * HD * 2);
  _Float16* o16   = (_Float16*)alloc((size_t)ROWS * DIM * 2);
  _Float16* h2    = (_Float16*)alloc((size_t)ROWS * DIM * 2);
  _Float16* g1b   = (_Float16*)alloc((size_t)ROWS * INTER * 2);
  _Float16* g3b   = (_Float16*)alloc((size_t)ROWS * INTER * 2);

  auto cast = [&](const float* src, _Float16* dst, int n) {
    cast_f32_f16<<<(n + 255) / 256, 256, 0, stream>>>(src, dst, n);
  };
  cast(Wq,  Wq16,  DIM * DIM);
  cast(Wkv, Wkv16, DIM * 512);
  cast(Wo,  Wo16,  DIM * DIM);
  cast(W1,  W1h,   DIM * INTER);
  cast(W3,  W3h,   DIM * INTER);
  cast(W2,  W2h,   INTER * DIM);

  // h = rmsnorm(x, w_attn_norm)
  rmsnorm_f16<<<ROWS, 256, 0, stream>>>(x, wan, h16);

  // q = h @ Wq ; kv = h @ Wkv
  gemm_wmma<0><<<dim3(DIM / 128, ROWS / 64), 256, 0, stream>>>(
      h16, Wq16, q32, nullptr, nullptr, ROWS, DIM, DIM);
  gemm_wmma<0><<<dim3(512 / 128, ROWS / 64), 256, 0, stream>>>(
      h16, Wkv16, kv32, nullptr, nullptr, ROWS, 512, DIM);

  // RoPE + layout transform
  rope_q<<<(BB * NN * NHEAD * 32) / 256, 256, 0, stream>>>(q32, rc, rsn, q16);
  rope_kv<<<(BB * NN * NKV * 32) / 256, 256, 0, stream>>>(kv32, rc, rsn, k16, v16);

  // attention
  flash_attn<<<dim3(BB * NHEAD, NN / 16), 32, 0, stream>>>(q16, k16, v16, o16);

  // out = x + o @ Wo
  gemm_wmma<1><<<dim3(DIM / 128, ROWS / 64), 256, 0, stream>>>(
      o16, Wo16, out, nullptr, x, ROWS, DIM, DIM);

  // h2 = rmsnorm(out, w_ffn_norm)
  rmsnorm_f16<<<ROWS, 256, 0, stream>>>(out, wfn, h2);

  // g1 = h2 @ W1 ; g3 = h2 @ W3  (f16 outputs)
  gemm_wmma<2><<<dim3(INTER / 128, ROWS / 64), 256, 0, stream>>>(
      h2, W1h, nullptr, g1b, nullptr, ROWS, INTER, DIM);
  gemm_wmma<2><<<dim3(INTER / 128, ROWS / 64), 256, 0, stream>>>(
      h2, W3h, nullptr, g3b, nullptr, ROWS, INTER, DIM);

  // gate = silu(g1) * g3
  swiglu_gate<<<(ROWS * INTER + 255) / 256, 256, 0, stream>>>(g1b, g3b, ROWS * INTER);

  // out = out + gate @ W2
  gemm_wmma<1><<<dim3(DIM / 128, ROWS / 64), 256, 0, stream>>>(
      g1b, W2h, out, nullptr, out, ROWS, DIM, INTER);
}
